// SE3VQVAE3DResNet_48498770707264
// MI455X (gfx1250) — matrix operand order repaired and checked by reading, not compiled, not run on hardware
//
#include <hip/hip_runtime.h>
#include <hip/hip_bf16.h>

typedef __attribute__((ext_vector_type(16))) _Float16 v16h;
typedef __attribute__((ext_vector_type(8)))  _Float16 v8h;
typedef __attribute__((ext_vector_type(8)))  float    v8f;

#define NROWS 16384   // B*L = 16*1024
#define LSEQ  1024
#define KCB   4096    // codebook size
#define DLAT  128     // latent dim

#define APAD 40       // A/B LDS row pitch (halves) for 32-k tiles: 80B = 20-bank stride
#define VPAD 136      // VQ LDS row pitch (halves) for 128-k rows: 272B stride
#define VBUF (64 * VPAD * 2)   // bytes per VQ B buffer (17408)

// ---- gfx1250 async global->LDS copy (16B per lane), ASYNCcnt tracked -------
__device__ __forceinline__ void async_copy16(void* lds_dst, const void* gsrc) {
  uint32_t l = (uint32_t)(uintptr_t)lds_dst;   // low 32 bits of generic ptr = LDS offset
  uint64_t g = (uint64_t)(uintptr_t)gsrc;
  asm volatile("global_load_async_to_lds_b128 %0, %1, off" :: "v"(l), "v"(g) : "memory");
}
__device__ __forceinline__ void async_wait0() {
  asm volatile("s_wait_asynccnt 0x0" ::: "memory");
}
__device__ __forceinline__ void async_wait3() {
  asm volatile("s_wait_asynccnt 0x3" ::: "memory");
}
__device__ __forceinline__ void async_wait4() {
  asm volatile("s_wait_asynccnt 0x4" ::: "memory");
}

__device__ __forceinline__ v16h frag16(const _Float16* lo, const _Float16* hi) {
  v8h a = *(const v8h*)lo;
  v8h b = *(const v8h*)hi;
  return __builtin_shufflevector(a, b, 0,1,2,3,4,5,6,7,8,9,10,11,12,13,14,15);
}

// ---------------------------------------------------------------------------
// Fused conv-as-GEMM:  Y[N,Cout] = sum_t Xshift_t[N,CinPad] * W[t][CinPad,Cout]
// All X buffers have row pitch Xstride (multiple of 32, zero padded); rows that
// fall outside a batch are async-copied from a zero page -> branch-free staging.
// Triple-buffered LDS, one barrier/K-step, async copies overlap WMMA compute.
// ---------------------------------------------------------------------------
__global__ __launch_bounds__(256) void gemm_conv(
    const _Float16* __restrict__ X, const _Float16* __restrict__ Wp,
    const float* __restrict__ scale, const float* __restrict__ shift,
    const _Float16* __restrict__ Res, _Float16* __restrict__ Yh,
    float* __restrict__ Yf, const _Float16* __restrict__ zp,
    int Xstride, int CinPad, int Cout, int CoutPad, int taps, int relu)
{
  __shared__ __attribute__((aligned(32))) _Float16 Asb[3][128 * APAD]; // 30 KB
  __shared__ __attribute__((aligned(32))) _Float16 Bsb[3][64 * APAD];  // 15 KB

  const int tid  = threadIdx.x;
  const int lane = tid & 31;
  const int wave = tid >> 5;
  const int wm   = wave & 3;   // row quadrant (32 rows)
  const int wn   = wave >> 2;  // col half (32 cols)
  const int hs   = lane >> 4;
  const int ml   = lane & 15;
  const int tileM = blockIdx.x;
  const int tileN = blockIdx.y;

  v8f acc[2][2] = {};
  const int nKb    = CinPad >> 5;
  const int half_t = taps >> 1;
  const int S      = taps * nKb;

  // per-thread staging coordinates
  const int ar = tid >> 1;            // A row 0..127
  const int ak = (tid & 1) * 16;      // A col-half within 32-k tile
  const int bc = tid >> 2;            // B row (output channel local) 0..63
  const int bp = (tid & 3) * 8;       // 16B part within 32-k row

  const int n  = tileM * 128 + ar;
  const int b0 = (n >> 10) << 10;     // batch start row

  auto stage = [&](int t, int kb, _Float16* Ad, _Float16* Bd) {
    int ns = n + t - half_t;
    bool rowok = (ns >= b0) && (ns < b0 + LSEQ);
    const _Float16* asrc = rowok ? (X + (size_t)ns * Xstride + (kb << 5) + ak) : zp;
    async_copy16(&Ad[ar * APAD + ak],     asrc);
    async_copy16(&Ad[ar * APAD + ak + 8], rowok ? asrc + 8 : zp);
    const _Float16* wb = Wp + ((size_t)(t * nKb + kb) * CoutPad + tileN * 64) * 32;
    async_copy16(&Bd[bc * APAD + bp], wb + bc * 32 + bp);
  };

  // prologue: stages 0 and 1 in flight
  stage(0, 0, Asb[0], Bsb[0]);
  if (S > 1) {
    int t1 = (nKb == 1) ? 1 : 0, kb1 = (nKb == 1) ? 0 : 1;
    stage(t1, kb1, Asb[1], Bsb[1]);
  }
  int t2 = (S > 2) ? (2 / nKb) : 0;
  int kb2 = (S > 2) ? (2 % nKb) : 0;

  int cb_ = 0;
  for (int s = 0; s < S; ++s) {
    if (s + 1 < S) async_wait3(); else async_wait0();   // stage s complete
    __syncthreads();
    if (s + 2 < S) {                                    // overlap copy of s+2
      int tgt = cb_ + 2; if (tgt >= 3) tgt -= 3;
      stage(t2, kb2, Asb[tgt], Bsb[tgt]);
      if (++kb2 == nKb) { kb2 = 0; ++t2; }
    }

    const _Float16* As = Asb[cb_];
    const _Float16* Bs = Bsb[cb_];
    v16h afr[2], bfr[2];
    #pragma unroll
    for (int tr = 0; tr < 2; ++tr) {
      const _Float16* ap = &As[(wm * 32 + tr * 16 + ml) * APAD];
      afr[tr] = frag16(ap + hs * 8, ap + 16 + hs * 8);
    }
    #pragma unroll
    for (int tc = 0; tc < 2; ++tc) {
      const _Float16* bpp = &Bs[(wn * 32 + tc * 16 + ml) * APAD + hs * 8];
      bfr[tc] = frag16(bpp, bpp + 16);
    }
    #pragma unroll
    for (int tr = 0; tr < 2; ++tr)
      #pragma unroll
      for (int tc = 0; tc < 2; ++tc)
        acc[tr][tc] = __builtin_amdgcn_wmma_f32_16x16x32_f16(
            false, afr[tr], false, bfr[tc], (short)0, acc[tr][tc], false, false);

    cb_ = (cb_ == 2) ? 0 : cb_ + 1;
  }

  // ---- fused epilogue: BN/bias + residual + ReLU + store ----
  #pragma unroll
  for (int tr = 0; tr < 2; ++tr) {
    #pragma unroll
    for (int tc = 0; tc < 2; ++tc) {
      int col = tileN * 64 + wn * 32 + tc * 16 + ml;
      if (col < Cout) {
        float sc = scale[col], sh = shift[col];
        #pragma unroll
        for (int r = 0; r < 8; ++r) {
          int row = tileM * 128 + wm * 32 + tr * 16 + hs * 8 + r; // C-matrix layout
          float v = acc[tr][tc][r] * sc + sh;
          if (Res)  v += (float)Res[(size_t)row * Cout + col];
          if (relu) v = fmaxf(v, 0.0f);
          if (Yh) Yh[(size_t)row * Cout + col] = (_Float16)v;
          if (Yf) Yf[(size_t)row * Cout + col] = v;
        }
      }
    }
  }
}

// ---------------------------------------------------------------------------
// VQ argmin: S = cnorm[j] - 2 * X(16384x128) . C^T, fused running min per row.
// A tile staged once; its LDS region is then reused for the triple-buffered
// codebook tiles (A lives in registers as fragments).
// ---------------------------------------------------------------------------
__global__ __launch_bounds__(256) void vq_argmin(
    const _Float16* __restrict__ X, const _Float16* __restrict__ CBH,
    const float* __restrict__ cnorm, int* __restrict__ idxOut)
{
  __shared__ __attribute__((aligned(32))) char smem[3 * VBUF]; // 52224 B

  const int tid  = threadIdx.x;
  const int lane = tid & 31;
  const int wave = tid >> 5;
  const int wm = wave & 3, wn = wave >> 2;
  const int hs = lane >> 4, ml = lane & 15;
  const int tileM = blockIdx.x;

  // stage X tile (128 rows x 128 k = 32KB) into front of smem
  {
    _Float16* As = (_Float16*)smem;
    const char* src = (const char*)(X + (size_t)tileM * 128 * DLAT);
    #pragma unroll
    for (int e = 0; e < 8; ++e) {
      int q   = tid + 256 * e;
      int row = q >> 4;
      int off = (q & 15) * 16;
      async_copy16((char*)As + row * (VPAD * 2) + off, src + row * 256 + off);
    }
  }
  async_wait0();
  __syncthreads();

  v16h afr[2][4];
  {
    const _Float16* As = (const _Float16*)smem;
    #pragma unroll
    for (int tr = 0; tr < 2; ++tr)
      #pragma unroll
      for (int kb = 0; kb < 4; ++kb) {
        const _Float16* ap = &As[(wm * 32 + tr * 16 + ml) * VPAD + kb * 32];
        afr[tr][kb] = frag16(ap + hs * 8, ap + 16 + hs * 8);
      }
  }
  __syncthreads();   // A region now dead -> reuse for B buffers

  auto stageB = [&](int chunk, int buf) {
    char* dst = smem + (size_t)buf * VBUF;
    const char* src = (const char*)(CBH + (size_t)chunk * 64 * DLAT);
    #pragma unroll
    for (int e = 0; e < 4; ++e) {
      int q   = tid + 256 * e;
      int row = q >> 4;
      int off = (q & 15) * 16;
      async_copy16(dst + row * (VPAD * 2) + off, src + row * 256 + off);
    }
  };

  float minv[2][8];
  int   mini[2][8];
  #pragma unroll
  for (int tr = 0; tr < 2; ++tr)
    #pragma unroll
    for (int r = 0; r < 8; ++r) { minv[tr][r] = 3.4e38f; mini[tr][r] = 0; }

  const int NC = KCB / 64;   // 64 chunks
  stageB(0, 0);
  stageB(1, 1);

  int cb_ = 0;
  for (int chunk = 0; chunk < NC; ++chunk) {
    if (chunk + 1 < NC) async_wait4(); else async_wait0();
    __syncthreads();
    if (chunk + 2 < NC) {
      int tgt = cb_ + 2; if (tgt >= 3) tgt -= 3;
      stageB(chunk + 2, tgt);
    }

    const _Float16* Bs = (const _Float16*)(smem + (size_t)cb_ * VBUF);
    v8f acc[2][2] = {};
    #pragma unroll
    for (int kb = 0; kb < 4; ++kb) {
      v16h bfr[2];
      #pragma unroll
      for (int tc = 0; tc < 2; ++tc) {
        const _Float16* bp = &Bs[(wn * 32 + tc * 16 + ml) * VPAD + kb * 32 + hs * 8];
        bfr[tc] = frag16(bp, bp + 16);
      }
      #pragma unroll
      for (int tr = 0; tr < 2; ++tr)
        #pragma unroll
        for (int tc = 0; tc < 2; ++tc)
          acc[tr][tc] = __builtin_amdgcn_wmma_f32_16x16x32_f16(
              false, afr[tr][kb], false, bfr[tc], (short)0, acc[tr][tc], false, false);
    }
    #pragma unroll
    for (int tr = 0; tr < 2; ++tr)
      #pragma unroll
      for (int tc = 0; tc < 2; ++tc) {
        int col = chunk * 64 + wn * 32 + tc * 16 + ml;
        float cn = cnorm[col];
        #pragma unroll
        for (int r = 0; r < 8; ++r) {
          float s = cn - 2.0f * acc[tr][tc][r];
          if (s < minv[tr][r]) { minv[tr][r] = s; mini[tr][r] = col; }
        }
      }
    cb_ = (cb_ == 2) ? 0 : cb_ + 1;
  }

  // cross-lane / cross-wave reduction (overlay LDS)
  __syncthreads();
  float* redV = (float*)smem;             // 16 KB
  int*   redI = (int*)(smem + 16384);     // 16 KB
  #pragma unroll
  for (int tr = 0; tr < 2; ++tr)
    #pragma unroll
    for (int r = 0; r < 8; ++r) {
      int rl = wm * 32 + tr * 16 + hs * 8 + r;
      redV[rl * 32 + wn * 16 + ml] = minv[tr][r];
      redI[rl * 32 + wn * 16 + ml] = mini[tr][r];
    }
  __syncthreads();
  if (tid < 128) {
    float best = redV[tid * 32];
    int   bi   = redI[tid * 32];
    for (int s = 1; s < 32; ++s) {
      float v = redV[tid * 32 + s];
      if (v < best) { best = v; bi = redI[tid * 32 + s]; }
    }
    idxOut[tileM * 128 + tid] = bi;
  }
}

// q = codebook[idx]; commit += mean((q-x)^2); indices -> float out
__global__ __launch_bounds__(128) void vq_gather_commit(
    const int* __restrict__ idx, const float* __restrict__ cb,
    const _Float16* __restrict__ Xenc, _Float16* __restrict__ Q,
    float* __restrict__ outIdx, float* __restrict__ commitAcc)
{
  int row = blockIdx.x;
  int t   = threadIdx.x;
  int j   = idx[row];
  float c = cb[(size_t)j * DLAT + t];
  float x = (float)Xenc[(size_t)row * DLAT + t];
  float d = c - x; d *= d;
  Q[(size_t)row * DLAT + t] = (_Float16)c;
  __shared__ float sred[128];
  sred[t] = d;
  __syncthreads();
  for (int s = 64; s > 0; s >>= 1) { if (t < s) sred[t] += sred[t + s]; __syncthreads(); }
  if (t == 0) {
    atomicAdd(commitAcc, sred[0] * (1.0f / ((float)NROWS * (float)DLAT)));
    outIdx[row] = (float)j;
  }
}

// ------------------------- small prep kernels ------------------------------
// input (N,12) f32 -> (N,32) f16 zero-padded
__global__ void f2h_pad_kernel(const float* __restrict__ x, _Float16* __restrict__ y) {
  int i = blockIdx.x * blockDim.x + threadIdx.x;   // over N*32
  if (i < NROWS * 32) {
    int c = i & 31, n_ = i >> 5;
    y[i] = (c < 12) ? (_Float16)x[n_ * 12 + c] : (_Float16)0.0f;
  }
}

// fp16 row-major codebook + per-code squared norm
__global__ void prep_cb(const float* __restrict__ cb, _Float16* __restrict__ cbh,
                        float* __restrict__ cnorm) {
  int j = blockIdx.x * blockDim.x + threadIdx.x;
  if (j < KCB) {
    float s = 0.f;
    for (int k = 0; k < DLAT; ++k) {
      float v = cb[(size_t)j * DLAT + k];
      s += v * v;
      cbh[(size_t)j * DLAT + k] = (_Float16)v;
    }
    cnorm[j] = s;
  }
}

// pack (Cout,Cin,taps) f32 -> [t][kb][c (CoutPad)][k (32)] f16, zero padded
__global__ void pack_w(const float* __restrict__ w, _Float16* __restrict__ wp,
                       int Cout, int Cin, int taps, int CinPad, int CoutPad) {
  size_t i = (size_t)blockIdx.x * blockDim.x + threadIdx.x;
  int nKb = CinPad >> 5;
  size_t tot = (size_t)taps * nKb * CoutPad * 32;
  if (i < tot) {
    int kl = (int)(i & 31);
    size_t r = i >> 5;
    int c  = (int)(r % CoutPad); r /= CoutPad;
    int kb = (int)(r % nKb);
    int t  = (int)(r / nKb);
    int ci = kb * 32 + kl;
    float v = 0.f;
    if (c < Cout && ci < Cin) v = w[((size_t)c * Cin + ci) * taps + t];
    wp[i] = (_Float16)v;
  }
}

// fold conv bias + BN:  y = acc*scale + shift
__global__ void make_ss(const float* __restrict__ b, const float* __restrict__ g,
                        const float* __restrict__ be, const float* __restrict__ m,
                        const float* __restrict__ v, float* __restrict__ scale,
                        float* __restrict__ shift, int C, int hasbn) {
  int i = blockIdx.x * blockDim.x + threadIdx.x;
  if (i < C) {
    if (hasbn) {
      float s = g[i] * rsqrtf(v[i] + 1e-5f);
      scale[i] = s;
      shift[i] = (b[i] - m[i]) * s + be[i];
    } else {
      scale[i] = 1.0f;
      shift[i] = b[i];
    }
  }
}

__global__ void zero_page(float* p, int nfloats) {
  int i = blockIdx.x * blockDim.x + threadIdx.x;
  if (i < nfloats) p[i] = 0.0f;
}

// ---------------------------------------------------------------------------
// Host orchestration. d_in: [0]=input_coords, then params leaves in JAX pytree
// order (dict keys sorted): codebook(1), dec_blocks(2..79), dec_head(80..85),
// dec_tail(86..99), enc_blocks(100..177), enc_head(178..191), enc_tail(192..197).
// ---------------------------------------------------------------------------
extern "C" void kernel_launch(void* const* d_in, const int* in_sizes, int n_in,
                              void* d_out, int out_size, void* d_ws, size_t ws_size,
                              hipStream_t stream) {
  (void)in_sizes; (void)n_in; (void)out_size; (void)ws_size;
  const float* inp = (const float*)d_in[0];
  auto P = [&](int i) -> const float* { return (const float*)d_in[i]; };

  char* ws = (char*)d_ws;
  size_t off = 0;
  auto alloc = [&](size_t bytes) -> char* {
    char* p = ws + off;
    off = (off + bytes + 255) & ~(size_t)255;
    return p;
  };
  _Float16* A0    = (_Float16*)alloc((size_t)NROWS * 512 * 2);
  _Float16* A1    = (_Float16*)alloc((size_t)NROWS * 512 * 2);
  _Float16* Xin   = (_Float16*)alloc((size_t)NROWS * 32 * 2);   // zero padded 12->32
  _Float16* CBH   = (_Float16*)alloc((size_t)KCB * DLAT * 2);
  float*    cnorm = (float*)alloc((size_t)KCB * 4);
  int*      idxb  = (int*)alloc((size_t)NROWS * 4);
  float*    sspool= (float*)alloc((size_t)38 * 1024 * 4);
  _Float16* zpage = (_Float16*)alloc(256);
  _Float16* wpool = (_Float16*)alloc((size_t)12 * 1024 * 1024 * 2);

  struct L { short wi, bi, bn, cin, cout, taps, relu, inb, outb, resb; };
  // inb/outb/resb: 0=A0, 1=A1, 2=Xin, 3=float d_out, -1=none
  static const L Ls[38] = {
    {197,196,192,  12,128,1,1, 2,0,-1},  // enc_tail conv+bn+relu
    {101,100, -1, 128,128,3,0, 0,1,-1},  // enc b0 entry conv
    {111,110,102, 128,128,3,1, 1,0,-1},  //   res1 conv1
    {113,112,106, 128,128,3,1, 0,1, 1},  //   res1 conv2 (+res,+relu)
    {123,122,114, 128,128,3,1, 1,0,-1},  //   res2 conv1
    {125,124,118, 128,128,3,1, 0,1, 1},  //   res2 conv2
    {127,126, -1, 128,320,3,0, 1,0,-1},  // enc b1 entry conv
    {137,136,128, 320,320,3,1, 0,1,-1},
    {139,138,132, 320,320,3,1, 1,0, 0},
    {149,148,140, 320,320,3,1, 0,1,-1},
    {151,150,144, 320,320,3,1, 1,0, 0},
    {153,152, -1, 320,512,3,0, 0,1,-1},  // enc b2 entry conv
    {163,162,154, 512,512,3,1, 1,0,-1},
    {165,164,158, 512,512,3,1, 0,1, 1},
    {175,174,166, 512,512,3,1, 1,0,-1},
    {177,176,170, 512,512,3,1, 0,1, 1},
    {187,186,178, 512,512,1,1, 1,0,-1},  // enc_head conv1
    {189,188,182, 512,512,1,1, 0,1,-1},  // enc_head conv2
    {191,190, -1, 512,128,1,0, 1,0,-1},  // enc_head conv3 -> Xenc in A0
    { 95, 94, -1, 128,512,1,0, 1,0,-1},  // dec_tail conv1 (input = Q in A1)
    { 97, 96, 86, 512,512,1,1, 0,1,-1},  // dec_tail conv2+bn1+relu
    { 99, 98, 90, 512,512,1,1, 1,0,-1},  // dec_tail conv3+bn2+relu
    { 13, 12,  4, 512,512,3,1, 0,1,-1},  // dec b0 res1 conv1
    { 15, 14,  8, 512,512,3,1, 1,0, 0},
    { 25, 24, 16, 512,512,3,1, 0,1,-1},
    { 27, 26, 20, 512,512,3,1, 1,0, 0},
    {  3,  2, -1, 512,320,3,0, 0,1,-1},  // dec b0 exit conv
    { 39, 38, 30, 320,320,3,1, 1,0,-1},
    { 41, 40, 34, 320,320,3,1, 0,1, 1},
    { 51, 50, 42, 320,320,3,1, 1,0,-1},
    { 53, 52, 46, 320,320,3,1, 0,1, 1},
    { 29, 28, -1, 320,128,3,0, 1,0,-1},  // dec b1 exit conv
    { 65, 64, 56, 128,128,3,1, 0,1,-1},
    { 67, 66, 60, 128,128,3,1, 1,0, 0},
    { 77, 76, 68, 128,128,3,1, 0,1,-1},
    { 79, 78, 72, 128,128,3,1, 1,0, 0},
    { 55, 54, -1, 128,128,3,0, 0,1,-1},  // dec b2 exit conv
    { 85, 84, 80, 128, 12,1,1, 1,3,-1},  // dec_head conv+bn+relu -> f32 out
  };

  float* out     = (float*)d_out;
  float* outIdx  = out + (size_t)NROWS * 12;
  float* commit  = out + (size_t)NROWS * 12 + NROWS;

  // prep
  f2h_pad_kernel<<<(NROWS * 32 + 255) / 256, 256, 0, stream>>>(inp, Xin);
  prep_cb<<<(KCB + 255) / 256, 256, 0, stream>>>(P(1), CBH, cnorm);
  zero_page<<<1, 64, 0, stream>>>((float*)zpage, 64);
  zero_page<<<1, 64, 0, stream>>>(commit, 1);

  size_t woff[38];
  size_t wcur = 0;
  for (int i = 0; i < 38; ++i) {
    const L& l = Ls[i];
    int cinp  = (l.cin + 31) & ~31;
    int coutp = (l.cout + 63) & ~63;
    woff[i] = wcur;
    size_t tot = (size_t)l.taps * cinp * coutp;
    wcur += tot;
    pack_w<<<(int)((tot + 255) / 256), 256, 0, stream>>>(
        P(l.wi), wpool + woff[i], l.cout, l.cin, l.taps, cinp, coutp);
    int hasbn = (l.bn >= 0) ? 1 : 0;
    int gi  = hasbn ? l.bn + 1 : l.bi;
    int bei = hasbn ? l.bn : l.bi;
    int mi  = hasbn ? l.bn + 2 : l.bi;
    int vi  = hasbn ? l.bn + 3 : l.bi;
    make_ss<<<(l.cout + 255) / 256, 256, 0, stream>>>(
        P(l.bi), P(gi), P(bei), P(mi), P(vi),
        sspool + (size_t)i * 1024, sspool + (size_t)i * 1024 + 512, l.cout, hasbn);
  }

  auto bufp = [&](int b) -> _Float16* {
    return b == 0 ? A0 : (b == 1 ? A1 : (b == 2 ? Xin : nullptr));
  };

  for (int i = 0; i < 38; ++i) {
    const L& l = Ls[i];
    int cinp  = (l.cin + 31) & ~31;
    int coutp = (l.cout + 63) & ~63;
    int xs    = (l.inb == 2) ? 32 : l.cin;   // X row pitch (padded input buffer)
    dim3 grid(NROWS / 128, coutp / 64);
    const _Float16* Xp = bufp(l.inb);
    _Float16* Yh = (l.outb == 3) ? nullptr : bufp(l.outb);
    float*    Yf = (l.outb == 3) ? out : nullptr;
    const _Float16* Rp = (l.resb < 0) ? nullptr : bufp(l.resb);
    gemm_conv<<<grid, 256, 0, stream>>>(
        Xp, wpool + woff[i], sspool + (size_t)i * 1024, sspool + (size_t)i * 1024 + 512,
        Rp, Yh, Yf, zpage, xs, cinp, l.cout, coutp, l.taps, l.relu);
    if (i == 18) {
      // VQ: Xenc in A0 -> indices; Q (decoder input) -> A1; commit loss -> d_out tail
      vq_argmin<<<dim3(NROWS / 128), 256, 0, stream>>>(A0, CBH, cnorm, idxb);
      vq_gather_commit<<<dim3(NROWS), 128, 0, stream>>>(idxb, P(1), A0, A1, outIdx, commit);
    }
  }
}
